// Node_Internal_Dv_Decoder_82429012345238
// MI455X (gfx1250) — matrix operand order repaired and checked by reading, not compiled, not run on hardware
//
#include <hip/hip_runtime.h>
#include <hip/hip_bf16.h>

typedef __attribute__((ext_vector_type(16))) __bf16 v16bf;
typedef __attribute__((ext_vector_type(8)))  float  v8f;

#define NN 100000
#define NE 3200000
#define LAT 128

// ---------------------------------------------------------------------------
// Kernel 1: zero the per-node accumulator region in workspace.
// Layout per node (stride 12 floats): [0..2]=force, [3..5]=torque,
// [6..8]=constraint-sum, [9]=count, [10..11]=pad.
// ---------------------------------------------------------------------------
__global__ void __launch_bounds__(256)
zero_acc_kernel(float* __restrict__ acc, int n) {
  int i = blockIdx.x * 256 + threadIdx.x;
  if (i < n) acc[i] = 0.0f;
}

// ---------------------------------------------------------------------------
// Kernel 2: edge -> node scatter with native f32 global atomics.
// ---------------------------------------------------------------------------
__global__ void __launch_bounds__(256)
edge_scatter_kernel(const int*   __restrict__ recv,
                    const float* __restrict__ ef,
                    const float* __restrict__ et,
                    const float* __restrict__ ec,
                    float*       __restrict__ acc) {
  int e = blockIdx.x * 256 + threadIdx.x;
  if (e >= NE) return;
  int r = recv[e];
  float* a = acc + (size_t)r * 12;
#pragma unroll
  for (int c = 0; c < 3; ++c) {
    __hip_atomic_fetch_add(a + c,     ef[(size_t)e * 3 + c], __ATOMIC_RELAXED, __HIP_MEMORY_SCOPE_AGENT);
    __hip_atomic_fetch_add(a + 3 + c, et[(size_t)e * 3 + c], __ATOMIC_RELAXED, __HIP_MEMORY_SCOPE_AGENT);
    __hip_atomic_fetch_add(a + 6 + c, ec[(size_t)e * 3 + c], __ATOMIC_RELAXED, __HIP_MEMORY_SCOPE_AGENT);
  }
  __hip_atomic_fetch_add(a + 9, 1.0f, __ATOMIC_RELAXED, __HIP_MEMORY_SCOPE_AGENT);
}

// ---------------------------------------------------------------------------
// Kernel 3: fused 4-MLP WMMA kernel + final combine.
// block = 256 threads = 8 waves; each wave owns a 16-node tile (128 nodes/WG).
// For each MLP: stage W1 (bf16, B-operand lane layout) in LDS, run 8 col-tiles
// x 4 k-slices of v_wmma_f32_16x16x32_bf16, then bias+ReLU+W2 dot with a
// 16-lane shfl_xor reduction. Lanes 0/16 hold per-row results and write out.
// ---------------------------------------------------------------------------
__global__ void __launch_bounds__(256)
node_mlp_kernel(const float* __restrict__ X,     // node_latent [NN,128]
                const float* __restrict__ vel,   // current_velocity [NN,3]
                const float* __restrict__ acc,   // scatter accumulators [NN,12]
                const float* __restrict__ mW1, const float* __restrict__ mb1,
                const float* __restrict__ mW2, const float* __restrict__ mb2,
                const float* __restrict__ iW1, const float* __restrict__ ib1,
                const float* __restrict__ iW2, const float* __restrict__ ib2,
                const float* __restrict__ eW1, const float* __restrict__ eb1,
                const float* __restrict__ eW2, const float* __restrict__ eb2,
                const float* __restrict__ vW1, const float* __restrict__ vb1,
                const float* __restrict__ vW2, const float* __restrict__ vb2,
                float* __restrict__ out) {
  // W1 in LDS, bf16, pre-swizzled into the WMMA B-operand lane layout:
  // slot(s,t,l) = ((s*8+t)*32 + l) holds 16 contiguous bf16 (32B) for lane l.
  __shared__ __bf16 sW1[LAT * LAT];   // 32 KB
  __shared__ float  sB1[LAT];
  __shared__ float  sW2[LAT * 3];     // padded to od=3
  __shared__ float  sB2[3];

  const int tid  = threadIdx.x;
  const int lane = tid & 31;
  const int wave = tid >> 5;
  const int half = lane >> 4;        // 0: lanes 0-15, 1: lanes 16-31
  const int lrow = lane & 15;
  const int tileBase = (blockIdx.x * 8 + wave) * 16;
  const bool active = (tileBase < NN);

  // ---- A tile: 16 rows x 128 K, bf16, ISA 16-bit A layout, loaded once ----
  v16bf a[4];
  if (active) {
    int m = tileBase + lrow;
    if (m >= NN) m = NN - 1;                 // clamp for partial tail tile
    const float* xr = X + (size_t)m * LAT;
#pragma unroll
    for (int s = 0; s < 4; ++s) {
      const int k0 = s * 32 + half * 8;      // e=0..7 -> K=k0+e ; e=8..15 -> K=k0+16+(e-8)
#pragma unroll
      for (int e = 0; e < 8; ++e) {
        a[s][e]     = (__bf16)xr[k0 + e];
        a[s][e + 8] = (__bf16)xr[k0 + 16 + e];
      }
    }
  }

  const float* W1s[4] = {mW1, iW1, eW1, vW1};
  const float* B1s[4] = {mb1, ib1, eb1, vb1};
  const float* W2s[4] = {mW2, iW2, eW2, vW2};
  const float* B2s[4] = {mb2, ib2, eb2, vb2};
  const int    ods[4] = {1, 1, 3, 1};

  // Per-row results for the 8 rows owned by this lane-half:
  // mlp 0 = inverse_mass, 1 = inverse_inertia, 2 = dv_ext(3), 3 = vscale
  float r_out[4][8][3];

#pragma unroll
  for (int mlp = 0; mlp < 4; ++mlp) {
    const int od = ods[mlp];

    __syncthreads();   // previous iteration's LDS reads are done
    // ---- stage W1 -> LDS (bf16, B layout). Coalesced global reads. ----
    const float* W1 = W1s[mlp];
    for (int idx = tid; idx < LAT * LAT; idx += 256) {
      const int k = idx >> 7;                // row of W1 (K dim)
      const int n = idx & 127;               // col of W1 (N dim)
      const int s  = k >> 5;                 // k-slice
      const int kk = k & 31;
      const int e  = (kk & 7) + ((kk >> 4) << 3); // element within lane
      const int lh = (kk >> 3) & 1;          // K%16<8 -> lanes 0-15, else 16-31
      const int l  = lh * 16 + (n & 15);
      const int t  = n >> 4;                 // col tile
      sW1[(((s * 8 + t) * 32) + l) * 16 + e] = (__bf16)W1[idx];
    }
    for (int idx = tid; idx < LAT; idx += 256) sB1[idx] = B1s[mlp][idx];
    for (int idx = tid; idx < LAT * 3; idx += 256) {
      const int n = idx / 3, o = idx % 3;
      sW2[idx] = (o < od) ? W2s[mlp][n * od + o] : 0.0f;
    }
    if (tid < 3) sB2[tid] = (tid < od) ? B2s[mlp][tid] : 0.0f;
    __syncthreads();

    if (active) {
      float partial[8][3];
#pragma unroll
      for (int v = 0; v < 8; ++v)
#pragma unroll
        for (int o = 0; o < 3; ++o) partial[v][o] = 0.0f;

#pragma unroll
      for (int t = 0; t < 8; ++t) {
        v8f c = {};
#pragma unroll
        for (int s = 0; s < 4; ++s) {
          const v16bf b = *(const v16bf*)(sW1 + (((s * 8 + t) * 32) + lane) * 16);
          c = __builtin_amdgcn_wmma_f32_16x16x32_bf16(
                  /*neg_a=*/false, a[s], /*neg_b=*/false, b,
                  /*c_mod=*/(short)0, c, /*reuse_a=*/false, /*reuse_b=*/false);
        }
        // c[v] = H[row = v + 8*half][col = t*16 + lrow]
        const int col = t * 16 + lrow;
        const float b1v = sB1[col];
        const float w0 = sW2[col * 3 + 0];
        const float w1 = sW2[col * 3 + 1];
        const float w2 = sW2[col * 3 + 2];
#pragma unroll
        for (int v = 0; v < 8; ++v) {
          float h = fmaxf(c[v] + b1v, 0.0f);   // ReLU(x@W1 + b1)
          partial[v][0] += h * w0;
          partial[v][1] += h * w1;
          partial[v][2] += h * w2;
        }
      }
      // reduce across the 16 lanes of each half (columns of H)
#pragma unroll
      for (int v = 0; v < 8; ++v)
#pragma unroll
        for (int o = 0; o < 3; ++o) {
          float p = partial[v][o];
          p += __shfl_xor(p, 1, 32);
          p += __shfl_xor(p, 2, 32);
          p += __shfl_xor(p, 4, 32);
          p += __shfl_xor(p, 8, 32);
          r_out[mlp][v][o] = p + sB2[o];
        }
    }
  }

  // ---- final combine: lanes 0 and 16 each own 8 rows ----
  if (active && lrow == 0) {
#pragma unroll
    for (int v = 0; v < 8; ++v) {
      const int row = tileBase + half * 8 + v;
      if (row < NN) {
        const float* ar = acc + (size_t)row * 12;
        float cnt = ar[9];
        cnt = fmaxf(cnt, 1.0f);
        const float inv_cnt = 1.0f / cnt;
        const float invm = r_out[0][v][0];
        const float invi = r_out[1][v][0];
        const float vsc  = r_out[3][v][0];
#pragma unroll
        for (int c = 0; c < 3; ++c) {
          const float dv   = invm * ar[c];
          const float dav  = invi * ar[3 + c];
          const float disp = (vel[(size_t)row * 3 + c] + r_out[2][v][c]) * vsc
                             + ar[6 + c] * inv_cnt;
          out[(size_t)row * 3 + c]                  = dv;    // delta_velocity
          out[(size_t)(NN + row) * 3 + c]           = dav;   // delta_angular_velocity
          out[(size_t)(2 * NN + row) * 3 + c]       = disp;  // displacement
        }
      }
    }
  }
}

// ---------------------------------------------------------------------------
extern "C" void kernel_launch(void* const* d_in, const int* in_sizes, int n_in,
                              void* d_out, int out_size, void* d_ws, size_t ws_size,
                              hipStream_t stream) {
  (void)in_sizes; (void)n_in; (void)out_size; (void)ws_size;

  const int*   edge_index = (const int*)d_in[0];
  const int*   recv       = edge_index + NE;          // edge_index[1]
  const float* node_lat   = (const float*)d_in[1];
  const float* vel        = (const float*)d_in[2];
  const float* ef         = (const float*)d_in[3];
  const float* et         = (const float*)d_in[4];
  const float* ec         = (const float*)d_in[5];
  const float* mW1 = (const float*)d_in[6],  *mb1 = (const float*)d_in[7];
  const float* mW2 = (const float*)d_in[8],  *mb2 = (const float*)d_in[9];
  const float* iW1 = (const float*)d_in[10], *ib1 = (const float*)d_in[11];
  const float* iW2 = (const float*)d_in[12], *ib2 = (const float*)d_in[13];
  const float* eW1 = (const float*)d_in[14], *eb1 = (const float*)d_in[15];
  const float* eW2 = (const float*)d_in[16], *eb2 = (const float*)d_in[17];
  const float* vW1 = (const float*)d_in[18], *vb1 = (const float*)d_in[19];
  const float* vW2 = (const float*)d_in[20], *vb2 = (const float*)d_in[21];

  float* acc = (float*)d_ws;          // NN*12 floats = 4.8 MB
  float* out = (float*)d_out;

  const int nz = NN * 12;
  zero_acc_kernel<<<(nz + 255) / 256, 256, 0, stream>>>(acc, nz);
  edge_scatter_kernel<<<(NE + 255) / 256, 256, 0, stream>>>(recv, ef, et, ec, acc);

  const int nblocks = (NN + 127) / 128;   // 128 nodes per workgroup (8 waves x 16)
  node_mlp_kernel<<<nblocks, 256, 0, stream>>>(
      node_lat, vel, acc,
      mW1, mb1, mW2, mb2,
      iW1, ib1, iW2, ib2,
      eW1, eb1, eW2, eb2,
      vW1, vb1, vW2, vb2,
      out);
}
